// InteractionLayer_42314017800870
// MI455X (gfx1250) — matrix operand order repaired and checked by reading, not compiled, not run on hardware
//
#include <hip/hip_runtime.h>

typedef __attribute__((ext_vector_type(16))) _Float16 v16h;
typedef __attribute__((ext_vector_type(8)))  _Float16 v8h;
typedef __attribute__((ext_vector_type(8)))  float    v8f;

#define HID   128
#define T_SEQ 32
#define BATCH 1024
#define PHOR  10
#define NOUT  12          // 2 * num_vehicles
#define BT    (BATCH * T_SEQ)
#define BN_EPS 1e-5f

// ---- workspace layout (bytes, all 256-aligned) ----
#define O_FC1W    0u                      // 128*128 f16 (BN-folded)
#define O_FC2W    32768u                  // 64*128  f16 (BN-folded, contiguous after fc1)
#define O_ENCWIH  49152u                  // 512*64  f16
#define O_ENCWHH  114688u                 // 512*128 f16
#define O_DECW    245760u                 // 512*128 f16  (w_ih + w_hh folded: x_in == h)
#define O_OUTW    376832u                 // 16*128  f16  (rows >= 12 zero-padded)
#define O_FC1B    380928u                 // 128 f32 (BN-folded)
#define O_FC2B    381440u                 // 64  f32 (BN-folded)
#define O_CONVOUT 381696u                 // BT*128 f16
#define O_FEATS   (381696u + (unsigned)BT * 128u * 2u)   // [T][B][64] f16

// ---------------- CDNA5 async memory->LDS helpers ----------------
// Generic pointers to LDS have addr[31:0] == LDS byte offset (aperture in high bits),
// so truncation yields the dsaddr the async ops want in their VDST VGPR.
__device__ __forceinline__ unsigned lds_off(const void* p) {
    return (unsigned)(unsigned long long)p;
}

#define ASYNC_B32(ldsoff, gaddr)                                                   \
    asm volatile("global_load_async_to_lds_b32 %0, %1, off" ::"v"(ldsoff),         \
                 "v"(gaddr) : "memory")
#define ASYNC_B64(ldsoff, gaddr)                                                   \
    asm volatile("global_load_async_to_lds_b64 %0, %1, off" ::"v"(ldsoff),         \
                 "v"(gaddr) : "memory")

__device__ __forceinline__ void async_wait0() {
#if __has_builtin(__builtin_amdgcn_s_wait_asynccnt)
    __builtin_amdgcn_s_wait_asynccnt(0);
#else
    asm volatile("s_wait_asynccnt 0" ::: "memory");
#endif
}

// ---------------- WMMA helpers ----------------
__device__ __forceinline__ v8f wmma_f16(v16h a, v16h b, v8f c) {
    // D = A(16x32 f16) * B(32x16 f16) + C(16x16 f32)
    return __builtin_amdgcn_wmma_f32_16x16x32_f16(false, a, false, b, (short)0, c,
                                                  false, false);
}

// A fragment (16x32 f16 tile): lane (m = lane%16, g = lane/16)
__device__ __forceinline__ v16h afrag(const _Float16* base, int m, int g,
                                      int rowlen, int kbase) {
    const v8h lo = *(const v8h*)(base + (long)m * rowlen + kbase + g * 8);
    const v8h hi = *(const v8h*)(base + (long)m * rowlen + kbase + 16 + g * 8);
    return __builtin_shufflevector(lo, hi, 0,1,2,3,4,5,6,7,8,9,10,11,12,13,14,15);
}

// B fragment (32x16 f16 tile) from row-major W[n][k]: lane holds K = kbase+g*8+i (i<8), kbase+16+g*8+(i-8)
__device__ __forceinline__ v16h bfrag(const _Float16* w, int n, int g,
                                      int rowlen, int kbase) {
    const v8h lo = *(const v8h*)(w + (long)n * rowlen + kbase + g * 8);
    const v8h hi = *(const v8h*)(w + (long)n * rowlen + kbase + 16 + g * 8);
    return __builtin_shufflevector(lo, hi, 0,1,2,3,4,5,6,7,8,9,10,11,12,13,14,15);
}

__device__ __forceinline__ float sigmf(float x) { return 1.0f / (1.0f + __expf(-x)); }

// ---------------- weight prep: f32 -> f16, BN fold, decoder fold ----------------
__global__ __launch_bounds__(256) void prep_kernel(
    const float* __restrict__ fc1_w, const float* __restrict__ fc1_b,
    const float* __restrict__ g1, const float* __restrict__ be1,
    const float* __restrict__ m1, const float* __restrict__ v1,
    const float* __restrict__ fc2_w, const float* __restrict__ fc2_b,
    const float* __restrict__ g2, const float* __restrict__ be2,
    const float* __restrict__ m2, const float* __restrict__ v2,
    const float* __restrict__ enc_w_ih, const float* __restrict__ enc_w_hh,
    const float* __restrict__ dec_w_ih, const float* __restrict__ dec_w_hh,
    const float* __restrict__ out_w, unsigned char* __restrict__ ws)
{
    int id = blockIdx.x * blockDim.x + threadIdx.x;
    _Float16* fc1wh  = (_Float16*)(ws + O_FC1W);
    _Float16* fc2wh  = (_Float16*)(ws + O_FC2W);
    _Float16* ewihh  = (_Float16*)(ws + O_ENCWIH);
    _Float16* ewhhh  = (_Float16*)(ws + O_ENCWHH);
    _Float16* dwh    = (_Float16*)(ws + O_DECW);
    _Float16* owh    = (_Float16*)(ws + O_OUTW);
    float*    fc1bf  = (float*)(ws + O_FC1B);
    float*    fc2bf  = (float*)(ws + O_FC2B);

    if (id < 16384) {           // fc1: fold BN1 scale into rows
        int n = id >> 7;
        float s = g1[n] * rsqrtf(v1[n] + BN_EPS);
        fc1wh[id] = (_Float16)(fc1_w[id] * s);
    }
    if (id < 8192) {            // fc2: fold BN2 scale
        int n = id >> 7;
        float s = g2[n] * rsqrtf(v2[n] + BN_EPS);
        fc2wh[id] = (_Float16)(fc2_w[id] * s);
    }
    if (id < 32768) ewihh[id] = (_Float16)enc_w_ih[id];
    if (id < 65536) ewhhh[id] = (_Float16)enc_w_hh[id];
    if (id < 65536) dwh[id]   = (_Float16)(dec_w_ih[id] + dec_w_hh[id]);
    if (id < 2048) {            // output head padded to 16 rows
        int n = id >> 7, k = id & 127;
        owh[id] = (_Float16)((n < NOUT) ? out_w[n * 128 + k] : 0.0f);
    }
    if (id < 128) {
        float s = g1[id] * rsqrtf(v1[id] + BN_EPS);
        fc1bf[id] = (fc1_b[id] - m1[id]) * s + be1[id];
    }
    if (id < 64) {
        float s = g2[id] * rsqrtf(v2[id] + BN_EPS);
        fc2bf[id] = (fc2_b[id] - m2[id]) * s + be2[id];
    }
}

// ---------------- conv1+pool+conv2+pool (scalar VALU, 16 threads/frame) ----------------
__global__ __launch_bounds__(128) void conv_kernel(
    const float* __restrict__ obs,
    const float* __restrict__ w1, const float* __restrict__ b1,
    const float* __restrict__ w2, const float* __restrict__ b2,
    _Float16* __restrict__ convout)
{
    __shared__ __align__(16) float in_[8][400];    // [frame][4*10*10]
    __shared__ __align__(16) float p1[8][16][25];  // pooled conv1: [frame][ch][5*5]
    int tid = threadIdx.x;
    int fl  = tid >> 4;     // frame slot 0..7
    int ln  = tid & 15;     // 0..15
    long frame = (long)blockIdx.x * 8 + fl;

    // async-stage the 1600B frame tile: 25 x b32 per thread, stride 16 floats,
    // IOFFSET applies to both the LDS and the global address (ISA 08 §4.4)
    {
        unsigned ldso = lds_off(&in_[fl][0]) + (unsigned)ln * 4u;
        unsigned long long ga = (unsigned long long)(obs + frame * 400 + ln);
        asm volatile(
            "global_load_async_to_lds_b32 %0, %1, off\n\t"
            "global_load_async_to_lds_b32 %0, %1, off offset:64\n\t"
            "global_load_async_to_lds_b32 %0, %1, off offset:128\n\t"
            "global_load_async_to_lds_b32 %0, %1, off offset:192\n\t"
            "global_load_async_to_lds_b32 %0, %1, off offset:256\n\t"
            "global_load_async_to_lds_b32 %0, %1, off offset:320\n\t"
            "global_load_async_to_lds_b32 %0, %1, off offset:384\n\t"
            "global_load_async_to_lds_b32 %0, %1, off offset:448\n\t"
            "global_load_async_to_lds_b32 %0, %1, off offset:512\n\t"
            "global_load_async_to_lds_b32 %0, %1, off offset:576\n\t"
            "global_load_async_to_lds_b32 %0, %1, off offset:640\n\t"
            "global_load_async_to_lds_b32 %0, %1, off offset:704\n\t"
            "global_load_async_to_lds_b32 %0, %1, off offset:768\n\t"
            "global_load_async_to_lds_b32 %0, %1, off offset:832\n\t"
            "global_load_async_to_lds_b32 %0, %1, off offset:896\n\t"
            "global_load_async_to_lds_b32 %0, %1, off offset:960\n\t"
            "global_load_async_to_lds_b32 %0, %1, off offset:1024\n\t"
            "global_load_async_to_lds_b32 %0, %1, off offset:1088\n\t"
            "global_load_async_to_lds_b32 %0, %1, off offset:1152\n\t"
            "global_load_async_to_lds_b32 %0, %1, off offset:1216\n\t"
            "global_load_async_to_lds_b32 %0, %1, off offset:1280\n\t"
            "global_load_async_to_lds_b32 %0, %1, off offset:1344\n\t"
            "global_load_async_to_lds_b32 %0, %1, off offset:1408\n\t"
            "global_load_async_to_lds_b32 %0, %1, off offset:1472\n\t"
            "global_load_async_to_lds_b32 %0, %1, off offset:1536"
            ::"v"(ldso), "v"(ga) : "memory");
    }
    async_wait0();
    __syncthreads();

    // conv1: out channel = ln, SAME pad, relu, maxpool(2,2) -> 5x5
    float wr[36];
#pragma unroll
    for (int i = 0; i < 36; ++i) wr[i] = w1[ln * 36 + i];
    float bias1 = b1[ln];
    for (int py = 0; py < 5; ++py)
        for (int px = 0; px < 5; ++px) {
            float mx = 0.0f;  // relu values are >= 0
#pragma unroll
            for (int dy = 0; dy < 2; ++dy)
#pragma unroll
                for (int dx = 0; dx < 2; ++dx) {
                    int oy = 2 * py + dy, ox = 2 * px + dx;
                    float acc = bias1;
#pragma unroll
                    for (int ic = 0; ic < 4; ++ic)
#pragma unroll
                        for (int ky = 0; ky < 3; ++ky) {
                            int iy = oy + ky - 1;
                            if (iy < 0 || iy > 9) continue;
#pragma unroll
                            for (int kx = 0; kx < 3; ++kx) {
                                int ix = ox + kx - 1;
                                if (ix < 0 || ix > 9) continue;
                                acc += wr[ic * 9 + ky * 3 + kx] *
                                       in_[fl][ic * 100 + iy * 10 + ix];
                            }
                        }
                    mx = fmaxf(mx, fmaxf(acc, 0.0f));
                }
            p1[fl][ln][py * 5 + px] = mx;
        }
    __syncthreads();

    // conv2: 2 out channels per thread; pool(2,2) needs conv positions 0..3 only
#pragma unroll
    for (int half = 0; half < 2; ++half) {
        int oc = ln * 2 + half;
        float b2v = b2[oc];
        float acc[16];
#pragma unroll
        for (int i = 0; i < 16; ++i) acc[i] = b2v;
        for (int ic = 0; ic < 16; ++ic) {
            float wk[9];
#pragma unroll
            for (int i = 0; i < 9; ++i) wk[i] = w2[oc * 144 + ic * 9 + i];
#pragma unroll
            for (int oy = 0; oy < 4; ++oy)
#pragma unroll
                for (int ox = 0; ox < 4; ++ox) {
                    float a = 0.0f;
#pragma unroll
                    for (int ky = 0; ky < 3; ++ky) {
                        int iy = oy + ky - 1;
                        if (iy < 0 || iy > 4) continue;
#pragma unroll
                        for (int kx = 0; kx < 3; ++kx) {
                            int ix = ox + kx - 1;
                            if (ix < 0 || ix > 4) continue;
                            a += wk[ky * 3 + kx] * p1[fl][ic][iy * 5 + ix];
                        }
                    }
                    acc[oy * 4 + ox] += a;
                }
        }
#pragma unroll
        for (int py = 0; py < 2; ++py)
#pragma unroll
            for (int px = 0; px < 2; ++px) {
                float mx = fmaxf(fmaxf(acc[(2*py)*4 + 2*px],     acc[(2*py)*4 + 2*px+1]),
                                 fmaxf(acc[(2*py+1)*4 + 2*px],   acc[(2*py+1)*4 + 2*px+1]));
                convout[frame * 128 + oc * 4 + py * 2 + px] = (_Float16)fmaxf(mx, 0.0f);
            }
    }
}

// ---------------- fc1(+BN+relu) -> fc2(+BN+relu), WMMA from LDS-resident weights ----------------
__global__ __launch_bounds__(256) void fc_kernel(
    const _Float16* __restrict__ convout,
    const _Float16* __restrict__ wsrc,      // fc1 (32KB) + fc2 (16KB), contiguous
    const float* __restrict__ fc1bf, const float* __restrict__ fc2bf,
    _Float16* __restrict__ feats)
{
    __shared__ __align__(32) _Float16 wlds[16384 + 8192];  // 48KB: fc1 then fc2
    __shared__ __align__(32) _Float16 mid[8][16 * 128];    // 32KB
    int tid = threadIdx.x, wid = tid >> 5, l = tid & 31;
    int n16 = l & 15, g = l >> 4;
    long mt = (long)blockIdx.x * 8 + wid;      // 16-frame tile id (0..2047)

    // async-stage all fc weights into LDS once per workgroup: 12 x b128 per thread
    {
        unsigned ldso = lds_off(&wlds[0]) + (unsigned)tid * 16u;
        unsigned long long ga = (unsigned long long)wsrc + (unsigned long long)tid * 16u;
        asm volatile(
            "global_load_async_to_lds_b128 %0, %1, off\n\t"
            "global_load_async_to_lds_b128 %0, %1, off offset:4096\n\t"
            "global_load_async_to_lds_b128 %0, %1, off offset:8192\n\t"
            "global_load_async_to_lds_b128 %0, %1, off offset:12288\n\t"
            "global_load_async_to_lds_b128 %0, %1, off offset:16384\n\t"
            "global_load_async_to_lds_b128 %0, %1, off offset:20480\n\t"
            "global_load_async_to_lds_b128 %0, %1, off offset:24576\n\t"
            "global_load_async_to_lds_b128 %0, %1, off offset:28672\n\t"
            "global_load_async_to_lds_b128 %0, %1, off offset:32768\n\t"
            "global_load_async_to_lds_b128 %0, %1, off offset:36864\n\t"
            "global_load_async_to_lds_b128 %0, %1, off offset:40960\n\t"
            "global_load_async_to_lds_b128 %0, %1, off offset:45056"
            ::"v"(ldso), "v"(ga) : "memory");
    }

    // overlap: A fragments from global while weights stream into LDS
    const _Float16* xbase = convout + mt * 16 * 128;
    v16h a[4];
#pragma unroll
    for (int kk = 0; kk < 4; ++kk) a[kk] = afrag(xbase, n16, g, 128, kk * 32);

    async_wait0();
    __syncthreads();

    const _Float16* fc1l = &wlds[0];
    const _Float16* fc2l = &wlds[16384];

    _Float16* midw = &mid[wid][0];
#pragma unroll
    for (int nt = 0; nt < 8; ++nt) {           // fc1: N=128
        float b = fc1bf[nt * 16 + n16];
        v8f acc = {b, b, b, b, b, b, b, b};
#pragma unroll
        for (int kk = 0; kk < 4; ++kk)
            acc = wmma_f16(a[kk], bfrag(fc1l, nt * 16 + n16, g, 128, kk * 32), acc);
#pragma unroll
        for (int r = 0; r < 8; ++r) {
            int m = (g ? 8 : 0) + r;
            midw[m * 128 + nt * 16 + n16] = (_Float16)fmaxf(acc[r], 0.0f);
        }
    }
    __syncthreads();

    v16h a2[4];
#pragma unroll
    for (int kk = 0; kk < 4; ++kk) a2[kk] = afrag(midw, n16, g, 128, kk * 32);

    long frame0 = mt * 16;
    long bidx   = frame0 >> 5;                 // batch row (constant per tile)
    int  tbase  = (int)(frame0 & 31);
#pragma unroll
    for (int nt = 0; nt < 4; ++nt) {           // fc2: N=64
        float b = fc2bf[nt * 16 + n16];
        v8f acc = {b, b, b, b, b, b, b, b};
#pragma unroll
        for (int kk = 0; kk < 4; ++kk)
            acc = wmma_f16(a2[kk], bfrag(fc2l, nt * 16 + n16, g, 128, kk * 32), acc);
#pragma unroll
        for (int r = 0; r < 8; ++r) {
            int m = (g ? 8 : 0) + r;
            int t = tbase + m;
            feats[((long)t * BATCH + bidx) * 64 + nt * 16 + n16] =
                (_Float16)fmaxf(acc[r], 0.0f);
        }
    }
}

// ---------------- persistent LSTM encoder (32 steps) + decoder (10 steps) + head ----------------
__global__ __launch_bounds__(256) void lstm_kernel(
    const _Float16* __restrict__ feats,
    const _Float16* __restrict__ wih, const _Float16* __restrict__ whh,
    const float* __restrict__ enc_b,
    const _Float16* __restrict__ wdec, const float* __restrict__ dec_b,
    const _Float16* __restrict__ wout, const float* __restrict__ out_b,
    float* __restrict__ out)
{
    __shared__ __align__(32) _Float16 xbuf[2][16 * 64];
    __shared__ __align__(32) _Float16 hbuf[2][16 * 128];
    int tid = threadIdx.x, wid = tid >> 5, l = tid & 31;
    int n16 = l & 15, g = l >> 4;
    int b0 = blockIdx.x * 16;

    // zero initial hidden state (hbuf[0])
    {
        unsigned long long* z = (unsigned long long*)&hbuf[0][0];
        z[tid] = 0ull;
        z[tid + 256] = 0ull;
    }

    // encoder weights resident in VGPRs (wave-private gate columns)
    v16h rwih[4][2], rwhh[4][4];
    float bgate[4];
#pragma unroll
    for (int gi = 0; gi < 4; ++gi) {
        int n = gi * 128 + wid * 16 + n16;
#pragma unroll
        for (int kk = 0; kk < 2; ++kk) rwih[gi][kk] = bfrag(wih, n, g, 64, kk * 32);
#pragma unroll
        for (int kk = 0; kk < 4; ++kk) rwhh[gi][kk] = bfrag(whh, n, g, 128, kk * 32);
        bgate[gi] = enc_b[n];
    }

    v8f creg = {0, 0, 0, 0, 0, 0, 0, 0};

    // ---- encoder: 32 sequential steps, h ping-pongs through LDS ----
    for (int t = 0; t < T_SEQ; ++t) {
        int pb = t & 1;
        // async-stage x_t tile (2KB) straight into LDS
        {
            unsigned ldso = lds_off(&xbuf[pb][0]) + (unsigned)tid * 8u;
            unsigned long long ga =
                (unsigned long long)(feats + ((long)t * BATCH + b0) * 64) +
                (unsigned long long)tid * 8u;
            ASYNC_B64(ldso, ga);
        }
        if (t + 1 < T_SEQ)
            __builtin_prefetch((const void*)(feats + ((long)(t + 1) * BATCH + b0) * 64 +
                                             tid * 4), 0, 1);
        async_wait0();
        __syncthreads();

        v16h ax0 = afrag(&xbuf[pb][0], n16, g, 64, 0);
        v16h ax1 = afrag(&xbuf[pb][0], n16, g, 64, 32);
        v16h ah[4];
#pragma unroll
        for (int kk = 0; kk < 4; ++kk) ah[kk] = afrag(&hbuf[pb][0], n16, g, 128, kk * 32);

        v8f acc[4];
#pragma unroll
        for (int gi = 0; gi < 4; ++gi) {
            float b = bgate[gi];
            v8f a0 = {b, b, b, b, b, b, b, b};
            a0 = wmma_f16(ax0, rwih[gi][0], a0);
            a0 = wmma_f16(ax1, rwih[gi][1], a0);
#pragma unroll
            for (int kk = 0; kk < 4; ++kk) a0 = wmma_f16(ah[kk], rwhh[gi][kk], a0);
            acc[gi] = a0;
        }

        _Float16* hw = &hbuf[pb ^ 1][0];
#pragma unroll
        for (int r = 0; r < 8; ++r) {
            float iv = sigmf(acc[0][r]);
            float fv = sigmf(acc[1][r]);
            float gv = tanhf(acc[2][r]);
            float ov = sigmf(acc[3][r]);
            float cn = fv * creg[r] + iv * gv;
            creg[r] = cn;
            float hn = ov * tanhf(cn);
            int m = (g ? 8 : 0) + r;
            hw[m * 128 + wid * 16 + n16] = (_Float16)hn;
        }
    }
    // h_n now in hbuf[0]; c_n in creg; decoder input x_in == h at every step.

    // decoder weights (w_ih + w_hh folded) + head weights
    v16h rwd[4][4];
    float bd[4];
#pragma unroll
    for (int gi = 0; gi < 4; ++gi) {
        int n = gi * 128 + wid * 16 + n16;
#pragma unroll
        for (int kk = 0; kk < 4; ++kk) rwd[gi][kk] = bfrag(wdec, n, g, 128, kk * 32);
        bd[gi] = dec_b[n];
    }
    v16h rwo[4];
    float ob = 0.0f;
    if (wid == 0) {
#pragma unroll
        for (int kk = 0; kk < 4; ++kk) rwo[kk] = bfrag(wout, n16, g, 128, kk * 32);
        ob = (n16 < NOUT) ? out_b[n16] : 0.0f;
    }
    __syncthreads();

    // ---- decoder: 10 steps ----
    for (int s = 0; s < PHOR; ++s) {
        int rp = s & 1;
        v16h ah[4];
#pragma unroll
        for (int kk = 0; kk < 4; ++kk) ah[kk] = afrag(&hbuf[rp][0], n16, g, 128, kk * 32);

        v8f acc[4];
#pragma unroll
        for (int gi = 0; gi < 4; ++gi) {
            float b = bd[gi];
            v8f a0 = {b, b, b, b, b, b, b, b};
#pragma unroll
            for (int kk = 0; kk < 4; ++kk) a0 = wmma_f16(ah[kk], rwd[gi][kk], a0);
            acc[gi] = a0;
        }

        _Float16* hw = &hbuf[rp ^ 1][0];
#pragma unroll
        for (int r = 0; r < 8; ++r) {
            float iv = sigmf(acc[0][r]);
            float fv = sigmf(acc[1][r]);
            float gv = tanhf(acc[2][r]);
            float ov = sigmf(acc[3][r]);
            float cn = fv * creg[r] + iv * gv;
            creg[r] = cn;
            float hn = ov * tanhf(cn);
            int m = (g ? 8 : 0) + r;
            hw[m * 128 + wid * 16 + n16] = (_Float16)hn;
        }
        __syncthreads();

        if (wid == 0) {   // output head on full new h (needs all waves' columns)
            v16h ah2[4];
#pragma unroll
            for (int kk = 0; kk < 4; ++kk)
                ah2[kk] = afrag(&hbuf[rp ^ 1][0], n16, g, 128, kk * 32);
            v8f po = {ob, ob, ob, ob, ob, ob, ob, ob};
#pragma unroll
            for (int kk = 0; kk < 4; ++kk) po = wmma_f16(ah2[kk], rwo[kk], po);
            if (n16 < NOUT) {
#pragma unroll
                for (int r = 0; r < 8; ++r) {
                    int m = (g ? 8 : 0) + r;
                    out[((long)(b0 + m) * PHOR + s) * NOUT + n16] = po[r];
                }
            }
        }
    }
}

// ---------------- host launcher ----------------
extern "C" void kernel_launch(void* const* d_in, const int* in_sizes, int n_in,
                              void* d_out, int out_size, void* d_ws, size_t ws_size,
                              hipStream_t stream)
{
    (void)in_sizes; (void)n_in; (void)out_size; (void)ws_size;
    const float* obs  = (const float*)d_in[0];
    const float* c1w  = (const float*)d_in[1];
    const float* c1b  = (const float*)d_in[2];
    const float* c2w  = (const float*)d_in[3];
    const float* c2b  = (const float*)d_in[4];
    const float* fc1w = (const float*)d_in[5];
    const float* fc1b = (const float*)d_in[6];
    const float* g1   = (const float*)d_in[7];
    const float* be1  = (const float*)d_in[8];
    const float* m1   = (const float*)d_in[9];
    const float* v1   = (const float*)d_in[10];
    const float* fc2w = (const float*)d_in[11];
    const float* fc2b = (const float*)d_in[12];
    const float* g2   = (const float*)d_in[13];
    const float* be2  = (const float*)d_in[14];
    const float* m2   = (const float*)d_in[15];
    const float* v2   = (const float*)d_in[16];
    const float* ewih = (const float*)d_in[17];
    const float* ewhh = (const float*)d_in[18];
    const float* eb   = (const float*)d_in[19];
    const float* dwih = (const float*)d_in[20];
    const float* dwhh = (const float*)d_in[21];
    const float* db   = (const float*)d_in[22];
    const float* ow   = (const float*)d_in[23];
    const float* ob   = (const float*)d_in[24];
    unsigned char* ws = (unsigned char*)d_ws;
    float* out = (float*)d_out;

    prep_kernel<<<256, 256, 0, stream>>>(fc1w, fc1b, g1, be1, m1, v1,
                                         fc2w, fc2b, g2, be2, m2, v2,
                                         ewih, ewhh, dwih, dwhh, ow, ws);
    conv_kernel<<<4096, 128, 0, stream>>>(obs, c1w, c1b, c2w, c2b,
                                          (_Float16*)(ws + O_CONVOUT));
    fc_kernel<<<256, 256, 0, stream>>>((const _Float16*)(ws + O_CONVOUT),
                                       (const _Float16*)(ws + O_FC1W),
                                       (const float*)(ws + O_FC1B),
                                       (const float*)(ws + O_FC2B),
                                       (_Float16*)(ws + O_FEATS));
    lstm_kernel<<<64, 256, 0, stream>>>((const _Float16*)(ws + O_FEATS),
                                        (const _Float16*)(ws + O_ENCWIH),
                                        (const _Float16*)(ws + O_ENCWHH), eb,
                                        (const _Float16*)(ws + O_DECW), db,
                                        (const _Float16*)(ws + O_OUTW), ob, out);
}
